// ACNN_35416300323392
// MI455X (gfx1250) — matrix operand order repaired and checked by reading, not compiled, not run on hardware
//
#include <hip/hip_runtime.h>
#include <math.h>

typedef __attribute__((ext_vector_type(16))) _Float16 v16h;
typedef __attribute__((ext_vector_type(8)))  _Float16 v8h;
typedef __attribute__((ext_vector_type(8)))  float    v8f;

#define B_    128
#define L_    512
#define LP_   514           // L + 2 zero-pad rows (one each end)
#define DW_   256
#define DP2_  128           // 2*DP
#define NF_   512
#define D_    384           // DW + 2*DP
#define DWIN_ 1152          // 3*D

// ---------------- WMMA fragment helpers (wave32, v_wmma_f32_16x16x32_f16) ----------------
// A (16x32 f16): lane m = lane&15; halves i<8 -> K = hi*8+i ; i>=8 -> K = 16+hi*8+(i-8)
__device__ __forceinline__ v16h load_a_frag(const _Float16* p) {
    // p already points at row base + k0 + hi*8
    v8h lo = *(const v8h*)(p);
    v8h hh = *(const v8h*)(p + 16);
    return __builtin_shufflevector(lo, hh, 0,1,2,3,4,5,6,7,8,9,10,11,12,13,14,15);
}
// B (32x16 f16) from Bt (N x K row-major): lane n = lane&15; halves i<8 -> K = hi*8+i ; i>=8 -> K = 16+hi*8+(i-8)
__device__ __forceinline__ v16h load_bt_frag(const _Float16* p) {
    v8h lo = *(const v8h*)(p);
    v8h hh = *(const v8h*)(p + 16);
    return __builtin_shufflevector(lo, hh, 0,1,2,3,4,5,6,7,8,9,10,11,12,13,14,15);
}

// ---------------- block reductions ----------------
__device__ __forceinline__ float block_max(float v, float* s) {
    int t = threadIdx.x;
    s[t] = v; __syncthreads();
    for (int o = blockDim.x >> 1; o > 0; o >>= 1) {
        if (t < o) s[t] = fmaxf(s[t], s[t + o]);
        __syncthreads();
    }
    float r = s[0]; __syncthreads();
    return r;
}
__device__ __forceinline__ float block_sum(float v, float* s) {
    int t = threadIdx.x;
    s[t] = v; __syncthreads();
    for (int o = blockDim.x >> 1; o > 0; o >>= 1) {
        if (t < o) s[t] += s[t + o];
        __syncthreads();
    }
    float r = s[0]; __syncthreads();
    return r;
}

// ---------------- stage 0: tiny precomputes ----------------
__global__ void prep_v_kernel(const float* We1, const float* We2,
                              const float* e1, const float* e2,
                              float* v1, float* v2) {
    int idx = blockIdx.x * blockDim.x + threadIdx.x;   // b*DW + d
    int b = idx >> 8, d = idx & (DW_ - 1);
    const float* w1 = We1 + (size_t)d * DW_;
    const float* w2 = We2 + (size_t)d * DW_;
    const float* p1 = e1 + (size_t)b * DW_;
    const float* p2 = e2 + (size_t)b * DW_;
    float s1 = 0.f, s2 = 0.f;
    for (int e = 0; e < DW_; ++e) { s1 += w1[e] * p1[e]; s2 += w2[e] * p2[e]; }
    v1[idx] = s1; v2[idx] = s2;
}

__global__ void score_kernel(const float* x, const float* v1, const float* v2,
                             float* A1, float* A2) {
    int idx = blockIdx.x * blockDim.x + threadIdx.x;   // b*L + l
    int b = idx >> 9;
    const float* xp = x + (size_t)idx * DW_;
    const float* p1 = v1 + (size_t)b * DW_;
    const float* p2 = v2 + (size_t)b * DW_;
    float s1 = 0.f, s2 = 0.f;
    #pragma unroll 4
    for (int d = 0; d < DW_; ++d) { float xv = xp[d]; s1 += xv * p1[d]; s2 += xv * p2[d]; }
    A1[idx] = s1; A2[idx] = s2;
}

__global__ void alpha_kernel(const float* A1, const float* A2, float* alpha) {
    __shared__ float s[256];
    int b = blockIdx.x, t = threadIdx.x;
    const float* a1 = A1 + (size_t)b * L_;
    const float* a2 = A2 + (size_t)b * L_;
    float x0 = a1[t], x1 = a1[t + 256];
    float m1 = block_max(fmaxf(x0, x1), s);
    float e0 = expf(x0 - m1), e1v = expf(x1 - m1);
    float s1 = block_sum(e0 + e1v, s);
    float y0 = a2[t], y1 = a2[t + 256];
    float m2 = block_max(fmaxf(y0, y1), s);
    float f0 = expf(y0 - m2), f1 = expf(y1 - m2);
    float s2 = block_sum(f0 + f1, s);
    alpha[(size_t)b * L_ + t]       = 0.5f * (e0 / s1 + f0 / s2);
    alpha[(size_t)b * L_ + t + 256] = 0.5f * (e1v / s1 + f1 / s2);
}

// ---------------- f16 operand staging ----------------
// xcatP: (B, L+2, D) f16, rows 0 and L+1 are zeros (the jnp.pad rows).
__global__ void xcat_kernel(const float* x, const float* posVec, _Float16* xcatP) {
    size_t idx = (size_t)blockIdx.x * blockDim.x + threadIdx.x; // ((b*(L+2))+r)*D + d
    int d = (int)(idx % D_);
    size_t br = idx / D_;
    int r = (int)(br % LP_);
    size_t b = br / LP_;
    float v = 0.f;
    if (r > 0 && r <= L_) {
        size_t bl = b * L_ + (size_t)(r - 1);
        v = (d < DW_) ? x[bl * DW_ + d] : posVec[bl * DP2_ + (d - DW_)];
    }
    xcatP[idx] = (_Float16)v;
}

__global__ void cast_convw_kernel(const float* w, _Float16* out) {
    size_t idx = (size_t)blockIdx.x * blockDim.x + threadIdx.x;
    out[idx] = (_Float16)w[idx];
}

// UrwT[r][f] = sum_t rel_w[r,t] * U[f,t]   (this IS the Bt layout for GEMM2)
__global__ void urwt_kernel(const float* rel_w, const float* U, _Float16* urwT) {
    int idx = blockIdx.x * blockDim.x + threadIdx.x; // r*NF + f
    int r = idx >> 9, f = idx & (NF_ - 1);
    const float* rw = rel_w + (size_t)r * NF_;
    const float* uf = U + (size_t)f * NF_;
    float s = 0.f;
    #pragma unroll 4
    for (int t = 0; t < NF_; ++t) s += rw[t] * uf[t];
    urwT[idx] = (_Float16)s;
}

__global__ void copy_relw_kernel(const float* rel_w, float* out) {
    int idx = blockIdx.x * blockDim.x + threadIdx.x;
    out[idx] = rel_w[idx];
}

__global__ void init_wo_kernel(float* wo) {
    int idx = blockIdx.x * blockDim.x + threadIdx.x;
    wo[idx] = 0.f;
}

// ---------------- GEMM1: conv = tanh(alpha * (w_concat @ conv_w^T) + b) ----------------
// Per batch: (L x DWIN) @ (DWIN x NF). The window dimension is the outer loop (w = 0..2);
// A rows come from the zero-padded xcatP so the inner loop is completely branchless.
__global__ void __launch_bounds__(128)
gemm_conv_kernel(const _Float16* __restrict__ xcatP, const _Float16* __restrict__ convw,
                 const float* __restrict__ alpha, const float* __restrict__ conv_b,
                 _Float16* __restrict__ convLF, _Float16* __restrict__ convFL) {
    int b  = blockIdx.z;
    int l0 = blockIdx.y * 64;
    int f0 = blockIdx.x * 64;
    int lane = threadIdx.x & 31, wave = threadIdx.x >> 5;
    int hi = lane >> 4, lm = lane & 15;
    int n = f0 + wave * 16 + lm;                  // filter column for this lane
    const _Float16* Ab = xcatP + (size_t)b * LP_ * D_;
    const _Float16* bbase = convw + (size_t)n * DWIN_ + hi * 8;
    __builtin_prefetch(bbase, 0, 3);              // one-shot prefetch of this lane's B panel
    v8f acc[4] = {};
    #pragma unroll
    for (int w = 0; w < 3; ++w) {
        // padded row index: (l - 1 + w) + 1 == l + w ; always in [0, L+1]
        const _Float16* arow0 = Ab + (size_t)(l0 +  0 + lm + w) * D_ + hi * 8;
        const _Float16* arow1 = Ab + (size_t)(l0 + 16 + lm + w) * D_ + hi * 8;
        const _Float16* arow2 = Ab + (size_t)(l0 + 32 + lm + w) * D_ + hi * 8;
        const _Float16* arow3 = Ab + (size_t)(l0 + 48 + lm + w) * D_ + hi * 8;
        const _Float16* brow  = bbase + w * D_;
        #pragma unroll 4
        for (int d0 = 0; d0 < D_; d0 += 32) {
            v16h bf  = load_bt_frag(brow + d0);
            v16h af0 = load_a_frag(arow0 + d0);
            v16h af1 = load_a_frag(arow1 + d0);
            v16h af2 = load_a_frag(arow2 + d0);
            v16h af3 = load_a_frag(arow3 + d0);
            acc[0] = __builtin_amdgcn_wmma_f32_16x16x32_f16(false, af0, false, bf, (short)0, acc[0], false, false);
            acc[1] = __builtin_amdgcn_wmma_f32_16x16x32_f16(false, af1, false, bf, (short)0, acc[1], false, false);
            acc[2] = __builtin_amdgcn_wmma_f32_16x16x32_f16(false, af2, false, bf, (short)0, acc[2], false, false);
            acc[3] = __builtin_amdgcn_wmma_f32_16x16x32_f16(false, af3, false, bf, (short)0, acc[3], false, false);
        }
    }
    float bias = conv_b[n];
    const float* al = alpha + (size_t)b * L_;
    #pragma unroll
    for (int mt = 0; mt < 4; ++mt) {
        #pragma unroll
        for (int j = 0; j < 8; ++j) {
            int l = l0 + mt * 16 + 8 * hi + j;     // D-row held by this lane/VGPR
            float val = tanhf(al[l] * acc[mt][j] + bias);
            _Float16 h = (_Float16)val;
            convLF[((size_t)b * L_ + l) * NF_ + n] = h;   // (b, l, f) for GEMM2 A
            convFL[((size_t)b * NF_ + n) * L_ + l] = h;   // (b, f, l) for GEMM3 A
        }
    }
}

// ---------------- GEMM2: G[b][r][l] = (conv^T @ Urw)^T, stored r-major ----------------
__global__ void __launch_bounds__(128)
gemm_attn_kernel(const _Float16* __restrict__ convLF, const _Float16* __restrict__ urwT,
                 float* __restrict__ G) {
    int b  = blockIdx.z;
    int l0 = blockIdx.y * 64;
    int r0 = blockIdx.x * 64;
    int lane = threadIdx.x & 31, wave = threadIdx.x >> 5;
    int hi = lane >> 4, lm = lane & 15;
    int n = r0 + wave * 16 + lm;
    const _Float16* Ab = convLF + (size_t)b * L_ * NF_;
    const _Float16* arow0 = Ab + (size_t)(l0 +  0 + lm) * NF_ + hi * 8;
    const _Float16* arow1 = Ab + (size_t)(l0 + 16 + lm) * NF_ + hi * 8;
    const _Float16* arow2 = Ab + (size_t)(l0 + 32 + lm) * NF_ + hi * 8;
    const _Float16* arow3 = Ab + (size_t)(l0 + 48 + lm) * NF_ + hi * 8;
    const _Float16* brow  = urwT + (size_t)n * NF_ + hi * 8;
    v8f acc[4] = {};
    #pragma unroll 4
    for (int k0 = 0; k0 < NF_; k0 += 32) {
        v16h bf  = load_bt_frag(brow + k0);
        v16h af0 = load_a_frag(arow0 + k0);
        v16h af1 = load_a_frag(arow1 + k0);
        v16h af2 = load_a_frag(arow2 + k0);
        v16h af3 = load_a_frag(arow3 + k0);
        acc[0] = __builtin_amdgcn_wmma_f32_16x16x32_f16(false, af0, false, bf, (short)0, acc[0], false, false);
        acc[1] = __builtin_amdgcn_wmma_f32_16x16x32_f16(false, af1, false, bf, (short)0, acc[1], false, false);
        acc[2] = __builtin_amdgcn_wmma_f32_16x16x32_f16(false, af2, false, bf, (short)0, acc[2], false, false);
        acc[3] = __builtin_amdgcn_wmma_f32_16x16x32_f16(false, af3, false, bf, (short)0, acc[3], false, false);
    }
    // transposed store -> contiguous 8 floats per lane per tile
    #pragma unroll
    for (int mt = 0; mt < 4; ++mt) {
        float* dst = G + ((size_t)b * NF_ + n) * L_ + l0 + mt * 16 + 8 * hi;
        #pragma unroll
        for (int j = 0; j < 8; ++j) dst[j] = acc[mt][j];
    }
}

// ---------------- softmax over l for each (b,r) row; emit f16 AP (b,r,l) ----------------
__global__ void softmax_ap_kernel(const float* __restrict__ G, _Float16* __restrict__ ap) {
    __shared__ float s[256];
    size_t row = blockIdx.x;                       // b*NF + r
    int t = threadIdx.x;
    const float* g = G + row * L_;
    float x0 = g[t], x1 = g[t + 256];
    float m = block_max(fmaxf(x0, x1), s);
    float e0 = expf(x0 - m), e1v = expf(x1 - m);
    float sm = block_sum(e0 + e1v, s);
    float inv = 1.0f / sm;
    ap[row * L_ + t]       = (_Float16)(e0 * inv);
    ap[row * L_ + t + 256] = (_Float16)(e1v * inv);
}

// ---------------- GEMM3: wo[b,f] = relu(max_r sum_l conv[f,l]*AP[r,l]) ----------------
// relu(max) == max(relu): clamp candidates >=0, then uint-bitpattern atomicMax (init 0).
__global__ void __launch_bounds__(128)
gemm_pool_kernel(const _Float16* __restrict__ convFL, const _Float16* __restrict__ ap,
                 float* __restrict__ wo) {
    int b  = blockIdx.z;
    int f0 = blockIdx.y * 64;
    int r0 = blockIdx.x * 64;
    int lane = threadIdx.x & 31, wave = threadIdx.x >> 5;
    int hi = lane >> 4, lm = lane & 15;
    int n = r0 + wave * 16 + lm;
    const _Float16* Ab = convFL + (size_t)b * NF_ * L_;
    const _Float16* arow0 = Ab + (size_t)(f0 +  0 + lm) * L_ + hi * 8;
    const _Float16* arow1 = Ab + (size_t)(f0 + 16 + lm) * L_ + hi * 8;
    const _Float16* arow2 = Ab + (size_t)(f0 + 32 + lm) * L_ + hi * 8;
    const _Float16* arow3 = Ab + (size_t)(f0 + 48 + lm) * L_ + hi * 8;
    const _Float16* brow  = ap + ((size_t)b * NF_ + n) * L_ + hi * 8;
    v8f acc[4] = {};
    #pragma unroll 4
    for (int k0 = 0; k0 < L_; k0 += 32) {
        v16h bf  = load_bt_frag(brow + k0);
        v16h af0 = load_a_frag(arow0 + k0);
        v16h af1 = load_a_frag(arow1 + k0);
        v16h af2 = load_a_frag(arow2 + k0);
        v16h af3 = load_a_frag(arow3 + k0);
        acc[0] = __builtin_amdgcn_wmma_f32_16x16x32_f16(false, af0, false, bf, (short)0, acc[0], false, false);
        acc[1] = __builtin_amdgcn_wmma_f32_16x16x32_f16(false, af1, false, bf, (short)0, acc[1], false, false);
        acc[2] = __builtin_amdgcn_wmma_f32_16x16x32_f16(false, af2, false, bf, (short)0, acc[2], false, false);
        acc[3] = __builtin_amdgcn_wmma_f32_16x16x32_f16(false, af3, false, bf, (short)0, acc[3], false, false);
    }
    #pragma unroll
    for (int mt = 0; mt < 4; ++mt) {
        #pragma unroll
        for (int j = 0; j < 8; ++j) {
            float v = fmaxf(acc[mt][j], 0.0f);
            v = fmaxf(v, __shfl_xor(v, 1, 32));    // reduce max over 16 r-columns
            v = fmaxf(v, __shfl_xor(v, 2, 32));
            v = fmaxf(v, __shfl_xor(v, 4, 32));
            v = fmaxf(v, __shfl_xor(v, 8, 32));
            if (lm == 0) {
                int f = f0 + mt * 16 + 8 * hi + j;
                atomicMax((unsigned int*)(wo + (size_t)b * NF_ + f), __float_as_uint(v));
            }
        }
    }
}

// ---------------- host launcher ----------------
extern "C" void kernel_launch(void* const* d_in, const int* in_sizes, int n_in,
                              void* d_out, int out_size, void* d_ws, size_t ws_size,
                              hipStream_t stream) {
    (void)in_sizes; (void)n_in; (void)out_size; (void)ws_size;
    const float* x      = (const float*)d_in[0];
    const float* e1     = (const float*)d_in[1];
    const float* e2     = (const float*)d_in[2];
    const float* posVec = (const float*)d_in[3];
    const float* We1    = (const float*)d_in[4];
    const float* We2    = (const float*)d_in[5];
    const float* U      = (const float*)d_in[6];
    const float* conv_w = (const float*)d_in[7];
    const float* conv_b = (const float*)d_in[8];
    const float* rel_w  = (const float*)d_in[9];

    float* out = (float*)d_out;   // [wo: B*NF] ++ [rel_w: NF*NF]

    char* wp = (char*)d_ws;
    auto take = [&](size_t bytes) { char* p = wp; wp += (bytes + 255) & ~(size_t)255; return p; };
    float*    v1      = (float*)   take((size_t)B_ * DW_ * 4);
    float*    v2      = (float*)   take((size_t)B_ * DW_ * 4);
    float*    A1      = (float*)   take((size_t)B_ * L_ * 4);
    float*    A2      = (float*)   take((size_t)B_ * L_ * 4);
    float*    alpha   = (float*)   take((size_t)B_ * L_ * 4);
    _Float16* xcatP   = (_Float16*)take((size_t)B_ * LP_ * D_ * 2 + 256);  // +slack
    _Float16* convw16 = (_Float16*)take((size_t)NF_ * DWIN_ * 2);
    _Float16* urwT    = (_Float16*)take((size_t)NF_ * NF_ * 2);
    _Float16* convLF  = (_Float16*)take((size_t)B_ * L_ * NF_ * 2);
    _Float16* convFL  = (_Float16*)take((size_t)B_ * NF_ * L_ * 2);
    float*    G       = (float*)   take((size_t)B_ * NF_ * L_ * 4);
    _Float16* ap      = (_Float16*)take((size_t)B_ * NF_ * L_ * 2);

    prep_v_kernel<<<(B_ * DW_) / 256, 256, 0, stream>>>(We1, We2, e1, e2, v1, v2);
    score_kernel<<<(B_ * L_) / 256, 256, 0, stream>>>(x, v1, v2, A1, A2);
    alpha_kernel<<<B_, 256, 0, stream>>>(A1, A2, alpha);
    xcat_kernel<<<(B_ * LP_ * D_) / 256, 256, 0, stream>>>(x, posVec, xcatP);
    cast_convw_kernel<<<(NF_ * DWIN_) / 256, 256, 0, stream>>>(conv_w, convw16);
    urwt_kernel<<<(NF_ * NF_) / 256, 256, 0, stream>>>(rel_w, U, urwT);
    copy_relw_kernel<<<(NF_ * NF_) / 256, 256, 0, stream>>>(rel_w, out + B_ * NF_);
    init_wo_kernel<<<(B_ * NF_) / 256, 256, 0, stream>>>(out);

    gemm_conv_kernel<<<dim3(NF_ / 64, L_ / 64, B_), 128, 0, stream>>>(
        xcatP, convw16, alpha, conv_b, convLF, convFL);
    gemm_attn_kernel<<<dim3(NF_ / 64, L_ / 64, B_), 128, 0, stream>>>(convLF, urwT, G);
    softmax_ap_kernel<<<B_ * NF_, 256, 0, stream>>>(G, ap);
    gemm_pool_kernel<<<dim3(NF_ / 64, NF_ / 64, B_), 128, 0, stream>>>(convFL, ap, out);
}
